// Attention_82343112999080
// MI455X (gfx1250) — matrix operand order repaired and checked by reading, not compiled, not run on hardware
//
#include <hip/hip_runtime.h>

typedef _Float16 h16 __attribute__((ext_vector_type(16)));
typedef _Float16 h8  __attribute__((ext_vector_type(8)));
typedef _Float16 h4  __attribute__((ext_vector_type(4)));
typedef _Float16 h2  __attribute__((ext_vector_type(2)));
typedef float    f8  __attribute__((ext_vector_type(8)));

#define WMMA_F32_F16(a, b, c) \
  __builtin_amdgcn_wmma_f32_16x16x32_f16(false, (a), false, (b), (short)0, (c), false, false)

constexpr int BATCH = 16;
constexpr int SEQ   = 2048;
constexpr int DIM   = 128;
constexpr int BR    = 64;          // query rows per workgroup (4 waves x 16)
constexpr int BC    = 64;          // key rows per LDS tile
constexpr int NT    = SEQ / BC;    // 32 key tiles
// fold 1/sqrt(128) * log2(e) into Q so softmax runs in exp2 domain
constexpr float QSCALE = 0.08838834764831845f * 1.4426950408889634f;

static __device__ __forceinline__ float fast_exp2(float x) {
#if __has_builtin(__builtin_amdgcn_exp2f)
  return __builtin_amdgcn_exp2f(x);
#else
  return exp2f(x);
#endif
}

static __device__ __forceinline__ float fast_rcp(float x) {
#if __has_builtin(__builtin_amdgcn_rcpf)
  return __builtin_amdgcn_rcpf(x);
#else
  return 1.0f / x;
#endif
}

__global__ __launch_bounds__(128) void fa_fwd_f16wmma(const float* __restrict__ Q,
                                                      const float* __restrict__ K,
                                                      const float* __restrict__ V,
                                                      float* __restrict__ O) {
  __shared__ _Float16 Ks[BC * DIM];          // keys, row-major f16              (16 KB)
  __shared__ _Float16 Vt[DIM * BC];          // values transposed f16: Vt[d][kr] (16 KB)
  __shared__ _Float16 Ps[4 * 16 * BC];       // per-wave P scratch               (8 KB)

  const int tid  = threadIdx.x;              // 0..127
  const int lane = tid & 31;
  const int wave = tid >> 5;                 // 0..3
  const int hs   = lane >> 4;                // half-wave select
  const int l15  = lane & 15;

  const int b  = blockIdx.x >> 5;            // 16 batches
  const int q0 = (blockIdx.x & 31) * BR;     // 32 query blocks of 64 rows

  // uniform per-batch base pointers (SGPRs) + 32-bit lane offsets -> GVS addressing
  const size_t bbase = (size_t)b * SEQ * DIM;
  const float* __restrict__ Kb = K + bbase;
  const float* __restrict__ Vb = V + bbase;
  float* __restrict__       Ob = O + bbase;

  // ---- Q A-fragments, pre-scaled into exp2 domain ----
  // A in contiguous k order (B uses the same): lanes 0-15 hold M=lane, K={0..15}; lanes 16-31 hold K={16..31}
  h16 aq[4];
  {
    const float* qrow = Q + bbase + (q0 + wave * 16 + l15) * DIM;
#pragma unroll
    for (int f = 0; f < 4; ++f) {
#pragma unroll
      for (int i = 0; i < 8; ++i) {
        aq[f][i]     = (_Float16)(qrow[f * 32 + hs * 16 + i] * QSCALE);
        aq[f][8 + i] = (_Float16)(qrow[f * 32 + hs * 16 + 8 + i] * QSCALE);
      }
    }
  }

  const f8 vzero = {};
  f8 o[8];
#pragma unroll
  for (int t = 0; t < 8; ++t) o[t] = vzero;
  float m[8], l[8];
#pragma unroll
  for (int r = 0; r < 8; ++r) { m[r] = -3.0e38f; l[r] = 0.0f; }

  _Float16* myP = Ps + wave * (16 * BC);

  for (int kt = 0; kt < NT; ++kt) {
    const int kr0 = kt * BC;

    // ---- cooperative staging, f32 -> f16 (32-bit offsets off uniform base) ----
    // K: row-major. 2048 float4s over 128 threads (16 each), fully coalesced.
#pragma unroll 4
    for (int i = 0; i < 16; ++i) {
      int e   = tid + i * 128;
      int row = e >> 5;
      int col = (e & 31) << 2;
      float4 kv = *(const float4*)(Kb + ((kr0 + row) * DIM + col));
      h4 hk = { (_Float16)kv.x, (_Float16)kv.y, (_Float16)kv.z, (_Float16)kv.w };
      *(h4*)&Ks[row * DIM + col] = hk;
    }
    // V: transposed, storing packed half2 (two key rows per b32 store).
#pragma unroll 4
    for (int i = 0; i < 8; ++i) {
      int p   = tid + i * 128;                // row-pair id, 0..1023
      int rp  = p >> 5;                       // rows 2rp, 2rp+1
      int col = (p & 31) << 2;
      int voff = (kr0 + 2 * rp) * DIM + col;
      float4 v0 = *(const float4*)(Vb + voff);
      float4 v1 = *(const float4*)(Vb + voff + DIM);
      h2 p0 = { (_Float16)v0.x, (_Float16)v1.x };
      h2 p1 = { (_Float16)v0.y, (_Float16)v1.y };
      h2 p2 = { (_Float16)v0.z, (_Float16)v1.z };
      h2 p3 = { (_Float16)v0.w, (_Float16)v1.w };
      *(h2*)&Vt[(col + 0) * BC + 2 * rp] = p0;
      *(h2*)&Vt[(col + 1) * BC + 2 * rp] = p1;
      *(h2*)&Vt[(col + 2) * BC + 2 * rp] = p2;
      *(h2*)&Vt[(col + 3) * BC + 2 * rp] = p3;
    }
    __syncthreads();

    // prefetch next tile toward the caches while this tile computes
    if (kt + 1 < NT) {
      const int noff = (kr0 + BC) * DIM + tid * 64;
      __builtin_prefetch(Kb + noff, 0, 0);
      __builtin_prefetch(Vb + noff, 0, 0);
    }

    // ---- S' = (Q K^T) in exp2 domain: 4 key sub-tiles, K-dim chained over 128 ----
    f8 s[4];
#pragma unroll
    for (int t = 0; t < 4; ++t) {
      f8 acc = vzero;
#pragma unroll
      for (int f = 0; f < 4; ++f) {
        h16 bk = *(const h16*)&Ks[(t * 16 + l15) * DIM + f * 32 + hs * 16];
        acc = WMMA_F32_F16(aq[f], bk, acc);
      }
      s[t] = acc;
    }

    // ---- online softmax (exp2 domain) ----
    float corr[8];
#pragma unroll
    for (int r = 0; r < 8; ++r) {
      float rm = fmaxf(fmaxf(s[0][r], s[1][r]), fmaxf(s[2][r], s[3][r]));
#pragma unroll
      for (int msk = 1; msk < 16; msk <<= 1)
        rm = fmaxf(rm, __shfl_xor(rm, msk, 32));
      float mn = fmaxf(m[r], rm);
      corr[r] = fast_exp2(m[r] - mn);
      m[r] = mn;
    }
    float rs[8];
#pragma unroll
    for (int r = 0; r < 8; ++r) rs[r] = 0.0f;
#pragma unroll
    for (int t = 0; t < 4; ++t) {
#pragma unroll
      for (int r = 0; r < 8; ++r) {
        float p = fast_exp2(s[t][r] - m[r]);
        s[t][r] = p;
        rs[r] += p;
      }
    }
#pragma unroll
    for (int r = 0; r < 8; ++r) {
#pragma unroll
      for (int msk = 1; msk < 16; msk <<= 1)
        rs[r] += __shfl_xor(rs[r], msk, 32);
      l[r] = l[r] * corr[r] + rs[r];
    }
#pragma unroll
    for (int t = 0; t < 8; ++t) {
#pragma unroll
      for (int r = 0; r < 8; ++r) o[t][r] *= corr[r];
    }

    // ---- P (f16) to per-wave LDS scratch, C-layout -> row-major 16x64 ----
#pragma unroll
    for (int t = 0; t < 4; ++t) {
#pragma unroll
      for (int r = 0; r < 8; ++r)
        myP[(hs * 8 + r) * BC + t * 16 + l15] = (_Float16)s[t][r];
    }
    __builtin_amdgcn_wave_barrier();  // same-wave LDS is in-order; block compiler reordering

    // ---- reload P as A-fragments (2 fragments of 16x32) ----
    h16 pa[2];
#pragma unroll
    for (int f = 0; f < 2; ++f) {
      const int base = l15 * BC + f * 32 + hs * 16;
      h8 lo = *(const h8*)&myP[base];
      h8 hi = *(const h8*)&myP[base + 8];
      pa[f] = __builtin_shufflevector(lo, hi, 0, 1, 2, 3, 4, 5, 6, 7,
                                      8, 9, 10, 11, 12, 13, 14, 15);
    }

    // ---- O += P V : 8 d-tiles, K-dim chained over 64 key rows ----
#pragma unroll
    for (int td = 0; td < 8; ++td) {
      f8 acc = o[td];
#pragma unroll
      for (int f = 0; f < 2; ++f) {
        h16 bv = *(const h16*)&Vt[(td * 16 + l15) * BC + f * 32 + hs * 16];
        acc = WMMA_F32_F16(pa[f], bv, acc);
      }
      o[td] = acc;
    }
    __syncthreads();
  }

  // ---- normalize and write out (fp32, 32-bit offsets off uniform base) ----
#pragma unroll
  for (int r = 0; r < 8; ++r) {
    float inv = fast_rcp(l[r]);
    const int rowoff = (q0 + wave * 16 + hs * 8 + r) * DIM;
#pragma unroll
    for (int td = 0; td < 8; ++td)
      Ob[rowoff + td * 16 + l15] = o[td][r] * inv;
  }
}

extern "C" void kernel_launch(void* const* d_in, const int* in_sizes, int n_in,
                              void* d_out, int out_size, void* d_ws, size_t ws_size,
                              hipStream_t stream) {
  (void)in_sizes; (void)n_in; (void)out_size; (void)d_ws; (void)ws_size;
  const float* Q = (const float*)d_in[0];
  const float* K = (const float*)d_in[1];
  const float* V = (const float*)d_in[2];
  float* O = (float*)d_out;

  dim3 grid(BATCH * (SEQ / BR));  // 512 workgroups
  dim3 block(128);                // 4 wave32 waves
  fa_fwd_f16wmma<<<grid, block, 0, stream>>>(Q, K, V, O);
}
